// MultiHeadAttention_4483945857347
// MI455X (gfx1250) — matrix operand order repaired and checked by reading, not compiled, not run on hardware
//
#include <hip/hip_runtime.h>
#include <hip/hip_bf16.h>

typedef unsigned short u16;
typedef unsigned int   u32;
typedef unsigned long long u64;

typedef float v8f  __attribute__((ext_vector_type(8)));
typedef u32   v8u  __attribute__((ext_vector_type(8)));
typedef __bf16 v16bf __attribute__((ext_vector_type(16)));

#define D_MODEL 1024
#define HEADS   16
#define DEPTH   64
#define SEQ     2048
#define BATCH   4

// ---------------- bf16 helpers (round-to-nearest-even pack) ----------------
__device__ __forceinline__ u16 f2bf(float f) {
    u32 a = __float_as_uint(f);
    a = (a + 0x7FFFu + ((a >> 16) & 1u)) >> 16;
    return (u16)a;
}
__device__ __forceinline__ u32 pack2bf(float lo, float hi) {
    return (u32)f2bf(lo) | ((u32)f2bf(hi) << 16);
}

// ---------------- WMMA fragment loaders (wave32 CDNA5 layouts) ----------------
// A 16x32 bf16: lane holds row M=lane&15; element e -> K = 16*(lane>>4) + e
__device__ __forceinline__ v16bf load_a_frag(const u16* rowp, int g) {
    const u32* q = (const u32*)rowp;
    v8u u;
#pragma unroll
    for (int p = 0; p < 8; ++p) u[p] = q[p + 8 * g];
    return __builtin_bit_cast(v16bf, u);
}
// B 32x16 bf16: lane holds col N=lane&15; element e -> K = 16*(lane>>4) + e.
__device__ __forceinline__ v16bf load_b_frag(const u16* p) {
    const u32* q = (const u32*)p;
    v8u u;
#pragma unroll
    for (int i = 0; i < 8; ++i) u[i] = q[i];
    return __builtin_bit_cast(v16bf, u);
}
__device__ __forceinline__ v8f wmma_bf16(v16bf a, v16bf b, v8f c) {
    return __builtin_amdgcn_wmma_f32_16x16x32_bf16(false, a, false, b,
                                                   (short)0, c, false, false);
}

// ---------------- CDNA5 async global->LDS copy (ASYNCcnt tracked) ----------------
// Each lane copies 32 bytes (2 x b128); offset applies to both LDS and global addr.
__device__ __forceinline__ void async_copy_tile_8k(u16* dst, const u16* src, int tid) {
    unsigned lds = (unsigned)(size_t)dst + (unsigned)tid * 32u;
    u64 gaddr = (u64)(size_t)src + (u64)tid * 32u;
    asm volatile(
        "global_load_async_to_lds_b128 %0, %1, off\n\t"
        "global_load_async_to_lds_b128 %0, %1, off offset:16"
        :: "v"(lds), "v"(gaddr) : "memory");
}
template <int N>
__device__ __forceinline__ void wait_async() {
    asm volatile("s_wait_asynccnt %0" :: "i"(N) : "memory");
}

// ==========================================================================
// GEMM: C[8192 x 1024] = (A(f32) x W(f32) + bias) * scale.
// 128x128 block tile, 8 waves as 4(M)x2(N), K-step 32.
// Double-buffered LDS, register-prefetched global loads (one barrier/iter).
// MODE 0: out bf16, split-head layout [B,H,S,64].  MODE 1: out f32, row-major.
// ==========================================================================
template <int MODE>
__global__ __launch_bounds__(256) void gemm128(const float* __restrict__ A,
                                               const float* __restrict__ W,
                                               const float* __restrict__ bias,
                                               float scale,
                                               u16* __restrict__ outb,
                                               float* __restrict__ outf) {
    __shared__ __align__(16) u16 lds_a[2][128 * 32];   // [row][k]
    __shared__ __align__(16) u16 lds_bT[2][128 * 32];  // [n][k]

    const int tid = threadIdx.x;
    const int wave = tid >> 5, lane = tid & 31;
    const int g = lane >> 4, ln = lane & 15;
    const int wm = wave >> 1, wn = wave & 1;
    const int bm = blockIdx.x * 128, bn = blockIdx.y * 128;

    v8f c[2][4] = {};

    const int arow = tid >> 1, akb = (tid & 1) * 16;
    const int brow = tid >> 3, bnb = (tid & 7) * 16;

    float4 a4[4], w4[4];
    auto load_regs = [&](int k0) {
        const float4* ap = (const float4*)(A + (size_t)(bm + arow) * D_MODEL + k0 + akb);
        const float4* wp = (const float4*)(W + (size_t)(k0 + brow) * D_MODEL + bn + bnb);
#pragma unroll
        for (int i = 0; i < 4; ++i) { a4[i] = ap[i]; w4[i] = wp[i]; }
    };
    auto store_tile = [&](int buf) {
        u32* dstA = (u32*)&lds_a[buf][arow * 32 + akb];
        const float* af = (const float*)a4;
        const float* wf = (const float*)w4;
#pragma unroll
        for (int i = 0; i < 8; ++i) dstA[i] = pack2bf(af[2 * i], af[2 * i + 1]);
#pragma unroll
        for (int i = 0; i < 16; ++i) lds_bT[buf][(bnb + i) * 32 + brow] = f2bf(wf[i]);
    };

    load_regs(0);
    store_tile(0);
    __syncthreads();

    for (int kt = 0; kt < D_MODEL / 32; ++kt) {
        const int cur = kt & 1;
        if (kt < D_MODEL / 32 - 1) load_regs((kt + 1) * 32);  // overlap with WMMA

        v16bf af[2], bfv[4];
#pragma unroll
        for (int fm = 0; fm < 2; ++fm)
            af[fm] = load_a_frag(&lds_a[cur][(wm * 32 + fm * 16 + ln) * 32], g);
#pragma unroll
        for (int fn = 0; fn < 4; ++fn)
            bfv[fn] = load_b_frag(&lds_bT[cur][(wn * 64 + fn * 16 + ln) * 32 + 16 * g]);
#pragma unroll
        for (int fm = 0; fm < 2; ++fm)
#pragma unroll
            for (int fn = 0; fn < 4; ++fn)
                c[fm][fn] = wmma_bf16(af[fm], bfv[fn], c[fm][fn]);

        if (kt < D_MODEL / 32 - 1) store_tile(1 - cur);
        __syncthreads();
    }

    // epilogue: C layout lane holds (M = r + 8g, N = ln)
#pragma unroll
    for (int fm = 0; fm < 2; ++fm) {
#pragma unroll
        for (int fn = 0; fn < 4; ++fn) {
            const int col = bn + wn * 64 + fn * 16 + ln;
            const float bv = bias[col];
#pragma unroll
            for (int r = 0; r < 8; ++r) {
                const int row = bm + wm * 32 + fm * 16 + r + 8 * g;
                const float val = (c[fm][fn][r] + bv) * scale;
                if (MODE == 0) {
                    const int b = row >> 11, s = row & (SEQ - 1);
                    const int h = col >> 6, d = col & (DEPTH - 1);
                    outb[(((size_t)(b * HEADS + h)) * SEQ + s) * DEPTH + d] = f2bf(val);
                } else {
                    outf[(size_t)row * D_MODEL + col] = val;
                }
            }
        }
    }
}

// ==========================================================================
// Attention.  Q pre-scaled by 1/sqrt(64) at projection time, so logits come
// straight out of the WMMA accumulators.  K tiles staged with async DMA into
// double-buffered LDS; V staged transposed via VGPRs (layout change needed).
// ==========================================================================
__device__ __forceinline__ void stage_vT(u16* dst, const u16* src, int tid) {
    const int key = tid & 63, dbase = (tid >> 6) * 16;
    const uint4* s = (const uint4*)(src + key * DEPTH + dbase);
    uint4 u0 = s[0], u1 = s[1];
    u32 uu[8] = {u0.x, u0.y, u0.z, u0.w, u1.x, u1.y, u1.z, u1.w};
#pragma unroll
    for (int i = 0; i < 8; ++i) {
        dst[(dbase + 2 * i) * 64 + key] = (u16)(uu[i] & 0xffffu);
        dst[(dbase + 2 * i + 1) * 64 + key] = (u16)(uu[i] >> 16);
    }
}
__device__ __forceinline__ v16bf load_q_frag(const u16* rowp, int g, int kstep) {
    const u32* q = (const u32*)rowp;
    const int base = 8 * g + 16 * kstep;
    v8u u;
#pragma unroll
    for (int p = 0; p < 8; ++p) u[p] = q[base + p];
    return __builtin_bit_cast(v16bf, u);
}

__global__ __launch_bounds__(256) void attention_kernel(
    const u16* __restrict__ qh, const u16* __restrict__ kh,
    const u16* __restrict__ vh, float* __restrict__ attn,
    float* __restrict__ ctxout) {
    __shared__ __align__(16) u16 lds_k[2][64 * 64];  // [key][d], double buffered
    __shared__ __align__(16) u16 lds_vT[64 * 64];    // [d][key]
    __shared__ __align__(16) u16 lds_p[8 * 16 * 64];

    const int tid = threadIdx.x;
    const int wave = tid >> 5, lane = tid & 31;
    const int g = lane >> 4, ln = lane & 15;
    const int bh = blockIdx.y;
    const int qbase = blockIdx.x * 128 + wave * 16;

    const u16* kbh = kh + (size_t)bh * SEQ * DEPTH;
    const u16* vbh = vh + (size_t)bh * SEQ * DEPTH;
    const u16* qp = qh + ((size_t)bh * SEQ + qbase + ln) * DEPTH;
    const v16bf aq0 = load_q_frag(qp, g, 0);
    const v16bf aq1 = load_q_frag(qp, g, 1);
    u16* pw = lds_p + wave * (16 * 64);

    // ---- sweep 1: row max ----
    float rmax[8];
#pragma unroll
    for (int r = 0; r < 8; ++r) rmax[r] = -3.402823466e38f;

    async_copy_tile_8k(&lds_k[0][0], kbh, tid);
    for (int kt = 0; kt < SEQ / 64; ++kt) {
        const u16* kcur = &lds_k[kt & 1][0];
        if (kt < SEQ / 64 - 1) {
            async_copy_tile_8k(&lds_k[(kt + 1) & 1][0],
                               kbh + (size_t)(kt + 1) * 64 * DEPTH, tid);
            wait_async<2>();
        } else {
            wait_async<0>();
        }
        __syncthreads();
#pragma unroll
        for (int nt = 0; nt < 4; ++nt) {
            v8f cacc = {};
            const u16* kr = &kcur[(nt * 16 + ln) * DEPTH + 16 * g];
            cacc = wmma_bf16(aq0, load_b_frag(kr), cacc);
            cacc = wmma_bf16(aq1, load_b_frag(kr + 32), cacc);
#pragma unroll
            for (int r = 0; r < 8; ++r) rmax[r] = fmaxf(rmax[r], cacc[r]);
        }
        __syncthreads();
    }
#pragma unroll
    for (int r = 0; r < 8; ++r)
#pragma unroll
        for (int m = 1; m < 16; m <<= 1)
            rmax[r] = fmaxf(rmax[r], __shfl_xor(rmax[r], m, 16));

    // ---- sweep 2: exp-sum and ctx = P @ V ----
    float rsum[8] = {};
    v8f ctx[4] = {};
    async_copy_tile_8k(&lds_k[0][0], kbh, tid);
    for (int kt = 0; kt < SEQ / 64; ++kt) {
        const u16* kcur = &lds_k[kt & 1][0];
        if (kt < SEQ / 64 - 1)
            async_copy_tile_8k(&lds_k[(kt + 1) & 1][0],
                               kbh + (size_t)(kt + 1) * 64 * DEPTH, tid);
        stage_vT(lds_vT, vbh + (size_t)kt * 64 * DEPTH, tid);
        if (kt < SEQ / 64 - 1) wait_async<2>(); else wait_async<0>();
        __syncthreads();
#pragma unroll
        for (int nt = 0; nt < 4; ++nt) {
            v8f cacc = {};
            const u16* kr = &kcur[(nt * 16 + ln) * DEPTH + 16 * g];
            cacc = wmma_bf16(aq0, load_b_frag(kr), cacc);
            cacc = wmma_bf16(aq1, load_b_frag(kr + 32), cacc);
#pragma unroll
            for (int r = 0; r < 8; ++r) {
                const float p = __expf(cacc[r] - rmax[r]);
                rsum[r] += p;
                pw[(r + 8 * g) * 64 + nt * 16 + ln] = f2bf(p);  // C -> A relayout
            }
        }
        const v16bf ap0 = load_a_frag(pw + ln * 64, g);
        const v16bf ap1 = load_a_frag(pw + ln * 64 + 32, g);
#pragma unroll
        for (int dn = 0; dn < 4; ++dn) {
            const u16* vr = &lds_vT[(dn * 16 + ln) * 64 + 16 * g];
            ctx[dn] = wmma_bf16(ap0, load_b_frag(vr), ctx[dn]);
            ctx[dn] = wmma_bf16(ap1, load_b_frag(vr + 32), ctx[dn]);
        }
        __syncthreads();
    }
#pragma unroll
    for (int r = 0; r < 8; ++r)
#pragma unroll
        for (int m = 1; m < 16; m <<= 1) rsum[r] += __shfl_xor(rsum[r], m, 16);
    float rinv[8];
#pragma unroll
    for (int r = 0; r < 8; ++r) rinv[r] = 1.0f / rsum[r];

    // write ctx in merged [B,S,1024] layout
    {
        const int b = bh >> 4, h = bh & (HEADS - 1);
#pragma unroll
        for (int dn = 0; dn < 4; ++dn)
#pragma unroll
            for (int r = 0; r < 8; ++r) {
                const int q = qbase + r + 8 * g;
                const int d = dn * 16 + ln;
                ctxout[((size_t)b * SEQ + q) * D_MODEL + h * DEPTH + d] =
                    ctx[dn][r] * rinv[r];
            }
    }

    // ---- sweep 3: recompute probabilities, stream normalized attn out ----
    async_copy_tile_8k(&lds_k[0][0], kbh, tid);
    for (int kt = 0; kt < SEQ / 64; ++kt) {
        const u16* kcur = &lds_k[kt & 1][0];
        if (kt < SEQ / 64 - 1) {
            async_copy_tile_8k(&lds_k[(kt + 1) & 1][0],
                               kbh + (size_t)(kt + 1) * 64 * DEPTH, tid);
            wait_async<2>();
        } else {
            wait_async<0>();
        }
        __syncthreads();
#pragma unroll
        for (int nt = 0; nt < 4; ++nt) {
            v8f cacc = {};
            const u16* kr = &kcur[(nt * 16 + ln) * DEPTH + 16 * g];
            cacc = wmma_bf16(aq0, load_b_frag(kr), cacc);
            cacc = wmma_bf16(aq1, load_b_frag(kr + 32), cacc);
#pragma unroll
            for (int r = 0; r < 8; ++r) {
                const float p = __expf(cacc[r] - rmax[r]) * rinv[r];
                const size_t q = (size_t)(qbase + r + 8 * g);
                attn[((size_t)bh * SEQ + q) * SEQ + kt * 64 + nt * 16 + ln] = p;
            }
        }
        __syncthreads();
    }
}

// ==========================================================================
extern "C" void kernel_launch(void* const* d_in, const int* in_sizes, int n_in,
                              void* d_out, int out_size, void* d_ws,
                              size_t ws_size, hipStream_t stream) {
    const float* v = (const float*)d_in[0];
    const float* k = (const float*)d_in[1];
    const float* q = (const float*)d_in[2];
    const float* wq = (const float*)d_in[3];
    const float* bq = (const float*)d_in[4];
    const float* wk = (const float*)d_in[5];
    const float* bk = (const float*)d_in[6];
    const float* wv = (const float*)d_in[7];
    const float* bv = (const float*)d_in[8];
    const float* wo = (const float*)d_in[9];
    const float* bo = (const float*)d_in[10];

    const size_t nElem = (size_t)BATCH * SEQ * D_MODEL;  // 8,388,608
    u16* qh = (u16*)d_ws;
    u16* khd = qh + nElem;
    u16* vhd = khd + nElem;
    float* ctx = (float*)(vhd + nElem);

    float* out = (float*)d_out;
    float* attn = out + nElem;  // attn follows `out` in the flat output

    const float qscale = 0.125f;  // 1/sqrt(DEPTH), folded into Q projection

    dim3 ggrid(BATCH * SEQ / 128, D_MODEL / 128);  // (64, 8)
    gemm128<0><<<ggrid, 256, 0, stream>>>(q, wq, bq, qscale, qh, nullptr);
    gemm128<0><<<ggrid, 256, 0, stream>>>(k, wk, bk, 1.0f, khd, nullptr);
    gemm128<0><<<ggrid, 256, 0, stream>>>(v, wv, bv, 1.0f, vhd, nullptr);

    dim3 agrid(SEQ / 128, BATCH * HEADS);  // (16, 64)
    attention_kernel<<<agrid, 256, 0, stream>>>(qh, khd, vhd, attn, ctx);

    gemm128<1><<<ggrid, 256, 0, stream>>>(ctx, wo, bo, 1.0f, nullptr, out);
}
